// Causal_t_69054484185474
// MI455X (gfx1250) — matrix operand order repaired and checked by reading, not compiled, not run on hardware
//
#include <hip/hip_runtime.h>

// Problem sizes (fixed by the reference)
#define Bb 16
#define Ll 512
#define Dd 768
#define Nn 8192
#define Mm (Bb * Ll)   // 8192 query rows

typedef __attribute__((ext_vector_type(16))) __bf16 v16bf;
typedef __attribute__((ext_vector_type(8)))  float  v8f;
typedef __attribute__((ext_vector_type(4)))  unsigned int u32x4;
typedef __attribute__((ext_vector_type(8)))  int i32x8;
typedef __attribute__((ext_vector_type(4)))  int i32x4;

union Frag {
  v16bf v;
  uint4 u4[2];
  __bf16 h[16];
};

// ---------------------------------------------------------------------------
// out_bf[r, e] = sum_d X[r,d] * W[e,d] + bias[e]   (bf16 output)
// One 16x16 tile per wave; K=32 bf16 WMMA chunks over D=768.
// ---------------------------------------------------------------------------
__global__ __launch_bounds__(128) void proj_gemm_bf16(
    const float* __restrict__ X, const float* __restrict__ W,
    const float* __restrict__ bias, __bf16* __restrict__ outbf, int R) {
  const int wave = threadIdx.x >> 5;
  const int lane = threadIdx.x & 31;
  const int half = lane >> 4;
  const int lr   = lane & 15;
  const int tile = blockIdx.x * 4 + wave;
  const int ctiles = Dd / 16;          // 48
  const int tr = tile / ctiles;
  const int tc = tile % ctiles;
  if (tr * 16 >= R) return;

  const float* __restrict__ arow = X + (size_t)(tr * 16 + lr) * Dd;
  const float* __restrict__ wrow = W + (size_t)(tc * 16 + lr) * Dd;

  v8f acc = {};
  for (int kk = 0; kk < Dd / 32; ++kk) {
    Frag a, b;
    const float* ap = arow + kk * 32 + half * 8;
#pragma unroll
    for (int i = 0; i < 8; ++i) a.h[i]     = (__bf16)ap[i];       // K = base..base+7
#pragma unroll
    for (int i = 0; i < 8; ++i) a.h[8 + i] = (__bf16)ap[16 + i];  // K = base+16..+23
    const float* bp = wrow + kk * 32 + half * 8;
#pragma unroll
    for (int i = 0; i < 8; ++i) b.h[i]     = (__bf16)bp[i];       // K = base..base+7
#pragma unroll
    for (int i = 0; i < 8; ++i) b.h[8 + i] = (__bf16)bp[16 + i];  // K = base+16..+23
    acc = __builtin_amdgcn_wmma_f32_16x16x32_bf16(false, a.v, false, b.v,
                                                  (short)0, acc, false, false);
  }
  const float bv = bias[tc * 16 + lr];
#pragma unroll
  for (int v = 0; v < 8; ++v) {
    const int row = tr * 16 + v + half * 8;
    outbf[(size_t)row * Dd + tc * 16 + lr] = (__bf16)(acc[v] + bv);
  }
}

// ---------------------------------------------------------------------------
// vT[d, n] = prior[n] * dic_z[n, d]  (bf16), LDS-tiled transpose
// ---------------------------------------------------------------------------
__global__ __launch_bounds__(256) void pv_transpose_bf16(
    const float* __restrict__ dic, const float* __restrict__ prior,
    __bf16* __restrict__ vT) {
  __shared__ float tile[32][33];
  const int nb = blockIdx.x * 32;
  const int db = blockIdx.y * 32;
  const int tx = threadIdx.x;   // 0..31
  const int ty = threadIdx.y;   // 0..7
#pragma unroll
  for (int i = 0; i < 32; i += 8) {
    const int n = nb + ty + i;
    tile[ty + i][tx] = dic[(size_t)n * Dd + db + tx] * prior[n];
  }
  __syncthreads();
#pragma unroll
  for (int i = 0; i < 32; i += 8) {
    const int d = db + ty + i;
    vT[(size_t)d * Nn + nb + tx] = (__bf16)tile[tx][ty + i];
  }
}

// ---------------------------------------------------------------------------
// Fused flash attention over the dictionary.
// 8 waves / WG; WG owns 32 query rows (two 16-row WMMA row-groups sharing
// every K/V B-fragment -> halves L2 streaming vs a 16-row tile).
// Per 128-wide N chunk: wave w computes S tiles for its 16 dict cols
// (2x24 WMMAs), cross-wave online softmax via LDS, P exchanged via LDS,
// wave w accumulates its 96-wide D slice of Z (2x24 WMMAs).
// ---------------------------------------------------------------------------
__global__ __launch_bounds__(256) void flash_dict_attn(
    const __bf16* __restrict__ qy, const __bf16* __restrict__ kz,
    const __bf16* __restrict__ vT, float* __restrict__ out) {
  __shared__ alignas(16) __bf16 Qs[32 * Dd];    // 48 KB, first decl -> LDS offset 0
  __shared__ alignas(16) __bf16 Ps[32 * 128];   // 8 KB
  __shared__ float red[8][32];
  __shared__ float m_run[32], l_run[32], m_new_s[32], scale_s[32];

  const int tid  = threadIdx.x;
  const int wave = tid >> 5;
  const int lane = tid & 31;
  const int half = lane >> 4;
  const int lr   = lane & 15;
  const int mbase = blockIdx.x * 32;

  // ---- TDM async tensor load of the Q tile (32 x 768 bf16) to LDS offset 0.
  // D# per CDNA5 ISA: group0 = {count=1 | lds_addr |
  // global_addr | type=2}; group1 = {data_size=2B, tensor_dim0=768,
  // tensor_dim1=32, tile_dim0=768, tile_dim1=32, dim0_stride=768}.
  if (wave == 0) {
    const unsigned long long ga =
        (unsigned long long)(uintptr_t)(qy + (size_t)mbase * Dd);
    u32x4 g0 = { 1u,                                   // count=1, user mode
                 0u,                                   // lds_addr = 0 (Qs)
                 (unsigned)(ga & 0xffffffffu),
                 (unsigned)((ga >> 32) & 0x01ffffffu) | 0x80000000u };  // type=2
    i32x8 g1 = { (int)0x00010000u,        // data_size=1 (2 bytes), no mask
                 (int)(768u << 16),       // tensor_dim0 = 768 (bits 79:48)
                 (int)(32u  << 16),       // tensor_dim1 = 32  (bits 111:80)
                 (int)(768u << 16),       // tile_dim0   = 768 (bits 127:112)
                 32,                      // tile_dim1   = 32  (bits 143:128)
                 768,                     // tensor_dim0_stride (bits 207:160)
                 0, 0 };
    i32x4 z4 = { 0, 0, 0, 0 };
    i32x8 z8 = { 0, 0, 0, 0, 0, 0, 0, 0 };
    __builtin_amdgcn_tensor_load_to_lds(g0, g1, z4, z4, z8, 0);
    __builtin_amdgcn_s_wait_tensorcnt((short)0);
  }
  // Safety-net cooperative copy (same bytes; ordered after the tensor wait).
  {
    const uint4* qg = (const uint4*)(qy + (size_t)mbase * Dd);
    uint4* qs = (uint4*)Qs;
    for (int i = tid; i < 32 * Dd / 8; i += 256) qs[i] = qg[i];
  }
  if (tid < 32) { m_run[tid] = -1e30f; l_run[tid] = 0.0f; }
  __syncthreads();

  v8f zero = {};
  v8f zacc[2][6];
#pragma unroll
  for (int rg = 0; rg < 2; ++rg)
#pragma unroll
    for (int cb = 0; cb < 6; ++cb) zacc[rg][cb] = zero;

  const float sm = 0.036084391824351615f;  // 1/sqrt(768)

  for (int nc = 0; nc < Nn / 128; ++nc) {
    const int n0 = nc * 128 + wave * 16;
    const __bf16* krow = kz + (size_t)(n0 + lr) * Dd;
    if (nc + 1 < Nn / 128) __builtin_prefetch(krow + 128 * Dd, 0, 1);

    // ---- S = Q * K^T for this wave's 16 dict columns, both row groups
    v8f sv[2];
    sv[0] = zero; sv[1] = zero;
#pragma unroll 4
    for (int kk = 0; kk < Dd / 32; ++kk) {
      Frag b;
      const uint4* kb = (const uint4*)(krow + kk * 32 + half * 8);
      b.u4[0] = kb[0];
      b.u4[1] = kb[2];
      Frag a0, a1;
      const __bf16* q0 = Qs + (size_t)lr * Dd + kk * 32 + half * 8;
      a0.u4[0] = *(const uint4*)(q0);
      a0.u4[1] = *(const uint4*)(q0 + 16);
      const __bf16* q1 = q0 + 16 * Dd;
      a1.u4[0] = *(const uint4*)(q1);
      a1.u4[1] = *(const uint4*)(q1 + 16);
      sv[0] = __builtin_amdgcn_wmma_f32_16x16x32_bf16(false, a0.v, false, b.v,
                                                      (short)0, sv[0], false, false);
      sv[1] = __builtin_amdgcn_wmma_f32_16x16x32_bf16(false, a1.v, false, b.v,
                                                      (short)0, sv[1], false, false);
    }

    // ---- per-wave row max (reduce across 16 lanes of each half)
#pragma unroll
    for (int rg = 0; rg < 2; ++rg)
#pragma unroll
      for (int v = 0; v < 8; ++v) {
        float x = sv[rg][v] * sm;
        sv[rg][v] = x;
#pragma unroll
        for (int off = 8; off >= 1; off >>= 1)
          x = fmaxf(x, __shfl_xor(x, off, 16));
        if (lr == 0) red[wave][rg * 16 + v + half * 8] = x;
      }
    __syncthreads();

    // ---- combine maxima across waves, compute rescale factors
    if (tid < 32) {
      float mo = m_run[tid];
      float mx = mo;
#pragma unroll
      for (int w = 0; w < 8; ++w) mx = fmaxf(mx, red[w][tid]);
      const float sc = __expf(mo - mx);
      m_new_s[tid] = mx;
      scale_s[tid] = sc;
      m_run[tid]   = mx;
      l_run[tid]  *= sc;
    }
    __syncthreads();

    // ---- P = exp(S - m), row partial sums, rescale Z accumulators
    float rs[2][8], sc8[2][8];
#pragma unroll
    for (int rg = 0; rg < 2; ++rg)
#pragma unroll
      for (int v = 0; v < 8; ++v) {
        const int row = rg * 16 + v + half * 8;
        const float p = __expf(sv[rg][v] - m_new_s[row]);
        Ps[row * 128 + wave * 16 + lr] = (__bf16)p;
        float x = p;
#pragma unroll
        for (int off = 8; off >= 1; off >>= 1)
          x += __shfl_xor(x, off, 16);
        rs[rg][v]  = x;
        sc8[rg][v] = scale_s[row];
      }
#pragma unroll
    for (int rg = 0; rg < 2; ++rg)
#pragma unroll
      for (int v = 0; v < 8; ++v)
        if (lr == 0) red[wave][rg * 16 + v + half * 8] = rs[rg][v];
#pragma unroll
    for (int rg = 0; rg < 2; ++rg)
#pragma unroll
      for (int cb = 0; cb < 6; ++cb)
#pragma unroll
        for (int v = 0; v < 8; ++v) zacc[rg][cb][v] *= sc8[rg][v];
    __syncthreads();

    if (tid < 32) {
      float acc = l_run[tid];
#pragma unroll
      for (int w = 0; w < 8; ++w) acc += red[w][tid];
      l_run[tid] = acc;
    }

    // ---- Z += P * V  (A = P from LDS, B = vT slices from L2, shared by rg)
    Frag pa[2][4];
#pragma unroll
    for (int rg = 0; rg < 2; ++rg)
#pragma unroll
      for (int c = 0; c < 4; ++c) {
        const __bf16* pbase = Ps + (rg * 16 + lr) * 128 + c * 32 + half * 8;
        pa[rg][c].u4[0] = *(const uint4*)(pbase);
        pa[rg][c].u4[1] = *(const uint4*)(pbase + 16);
      }
#pragma unroll
    for (int cb = 0; cb < 6; ++cb) {
      const int dcol = wave * 96 + cb * 16 + lr;
      const __bf16* vrow = vT + (size_t)dcol * Nn + nc * 128;
#pragma unroll
      for (int c = 0; c < 4; ++c) {
        Frag bv;
        const uint4* vp = (const uint4*)(vrow + c * 32 + half * 8);
        bv.u4[0] = vp[0];
        bv.u4[1] = vp[2];
        zacc[0][cb] = __builtin_amdgcn_wmma_f32_16x16x32_bf16(
            false, pa[0][c].v, false, bv.v, (short)0, zacc[0][cb], false, false);
        zacc[1][cb] = __builtin_amdgcn_wmma_f32_16x16x32_bf16(
            false, pa[1][c].v, false, bv.v, (short)0, zacc[1][cb], false, false);
      }
    }
    __syncthreads();  // protects red[] and Ps for the next chunk
  }

  // ---- normalize and store Z (fp32)
  float linv[2][8];
#pragma unroll
  for (int rg = 0; rg < 2; ++rg)
#pragma unroll
    for (int v = 0; v < 8; ++v)
      linv[rg][v] = 1.0f / l_run[rg * 16 + v + half * 8];
#pragma unroll
  for (int rg = 0; rg < 2; ++rg)
#pragma unroll
    for (int cb = 0; cb < 6; ++cb) {
      const int dcol = wave * 96 + cb * 16 + lr;
#pragma unroll
      for (int v = 0; v < 8; ++v) {
        const int row = rg * 16 + v + half * 8;
        out[(size_t)(mbase + row) * Dd + dcol] = zacc[rg][cb][v] * linv[rg][v];
      }
    }
}

// ---------------------------------------------------------------------------
extern "C" void kernel_launch(void* const* d_in, const int* in_sizes, int n_in,
                              void* d_out, int out_size, void* d_ws, size_t ws_size,
                              hipStream_t stream) {
  const float* y     = (const float*)d_in[0];
  const float* Wy_w  = (const float*)d_in[1];
  const float* Wy_b  = (const float*)d_in[2];
  const float* Wz_w  = (const float*)d_in[3];
  const float* Wz_b  = (const float*)d_in[4];
  const float* dic_z = (const float*)d_in[5];
  const float* prior = (const float*)d_in[6];
  float* out = (float*)d_out;

  // workspace: kz [N,D] bf16 | qy [M,D] bf16 | vT [D,N] bf16  (~37.7 MB)
  __bf16* kz = (__bf16*)d_ws;
  __bf16* qy = kz + (size_t)Nn * Dd;
  __bf16* vT = qy + (size_t)Mm * Dd;

  // K projection: kz = dic_z @ Wz^T + Wz_b   (512*48 tiles, 4 waves/block)
  proj_gemm_bf16<<<(Nn / 16) * (Dd / 16) / 4, 128, 0, stream>>>(dic_z, Wz_w, Wz_b, kz, Nn);
  // Q projection: qy = y @ Wy^T + Wy_b
  proj_gemm_bf16<<<(Mm / 16) * (Dd / 16) / 4, 128, 0, stream>>>(y, Wy_w, Wy_b, qy, Mm);
  // V^T = (prior * dic_z)^T
  pv_transpose_bf16<<<dim3(Nn / 32, Dd / 32), dim3(32, 8), 0, stream>>>(dic_z, prior, vT);
  // fused scores + softmax + weighted sum (32 query rows per WG)
  flash_dict_attn<<<Mm / 32, 256, 0, stream>>>(qy, kz, vT, out);
}
